// HeadAttn_59210419142956
// MI455X (gfx1250) — matrix operand order repaired and checked by reading, not compiled, not run on hardware
//
#include <hip/hip_runtime.h>

// ---------------------------------------------------------------------------
// Problem constants (must match reference)
// ---------------------------------------------------------------------------
constexpr int Bn    = 8;
constexpr int Ln    = 1024;
constexpr int Dn    = 256;
constexpr int Hn    = 8;
constexpr int NLn   = 2;
constexpr int PASTn = 256;
constexpr int FUTn  = 4;
constexpr int DHn   = Dn / Hn;            // 32
constexpr int Mrows  = Bn * Ln;           // 8192
constexpr int M2rows = Bn * (Ln - PASTn); // 6144
constexpr int NOUT   = Dn * FUTn;         // 1024

typedef __attribute__((ext_vector_type(16))) __bf16          v16bf;
typedef __attribute__((ext_vector_type(16))) unsigned short  v16u;
typedef __attribute__((ext_vector_type(8)))  unsigned int    v8u;
typedef __attribute__((ext_vector_type(8)))  float           v8f;
typedef __attribute__((ext_vector_type(4)))  float           v4f;   // native vec for NT stores

#define DEV static __device__ __forceinline__

// float -> bf16 (round-to-nearest-even), bit-level
DEV unsigned short f2bf(float f) {
    unsigned int u = __builtin_bit_cast(unsigned int, f);
    unsigned int r = (u + 0x7FFFu + ((u >> 16) & 1u)) >> 16;
    return (unsigned short)r;
}
DEV unsigned int f2bf2(float a, float b) {  // pack two bf16 into u32
    return (unsigned int)f2bf(a) | ((unsigned int)f2bf(b) << 16);
}
DEV v16bf asbf(v16u v)  { return __builtin_bit_cast(v16bf, v); }
DEV v16bf asbf8(v8u v)  { return __builtin_bit_cast(v16bf, v); }

// ---------------------------------------------------------------------------
// Generic WMMA GEMM:  C[M,N] = A[M,K] @ W[K,N] + bias (+ epilogue per MODE)
//   MODE 0: C -> bf16 (ushort) out                       (QKV projections)
//   MODE 1: C -> f32 out = acc + bias + res[m*N+n]       (O-proj, pre-LN)
//   MODE 2: C -> f32 out = acc + bias + res[arow*D + n%D], A rows sliced
// Block: 128 threads (4 waves); tile 16(M) x 64(N); K step 32.
// LDS tiles held as packed bf16 pairs (u32) to halve ds op count.
// ---------------------------------------------------------------------------
template <int MODE>
__global__ __launch_bounds__(128)
void wmma_gemm(const float* __restrict__ A, const float* __restrict__ W,
               const float* __restrict__ bias, const float* __restrict__ res,
               void* __restrict__ out, int M, int K, int N)
{
    __shared__ unsigned int sA[16 * 32 / 2];   // 16x32 bf16, u32-packed
    __shared__ unsigned int sW[32 * 64 / 2];   // 32x64 bf16, u32-packed

    const int tid  = threadIdx.x;
    const int wave = tid >> 5;
    const int lane = tid & 31;
    const int m0 = blockIdx.y * 16;
    const int n0 = blockIdx.x * 64;

    auto arow = [&](int r) -> int {
        int m = m0 + r;
        if (MODE == 2) {
            int b = m / (Ln - PASTn);
            int t = m - b * (Ln - PASTn);
            return b * Ln + PASTn + t;
        }
        return m;
    };

    v8f acc = {};
    for (int kk = 0; kk < K; kk += 32) {
        // ---- stage A tile: 16x32 f32 -> packed bf16 (4 floats / thread) ----
        {
            int r = tid >> 3;           // 0..15
            int c = (tid & 7) * 4;      // 0..28
            const float4 f = *(const float4*)(A + (size_t)arow(r) * K + kk + c);
            const int idx = (r * 32 + c) >> 1;
            sA[idx + 0] = f2bf2(f.x, f.y);
            sA[idx + 1] = f2bf2(f.z, f.w);
        }
        // ---- stage W tile: 32x64 f32 -> packed bf16 (16 floats / thread) ----
        {
            int r = tid >> 2;           // 0..31
            int c = (tid & 3) * 16;     // 0,16,32,48
            const float* wp = W + (size_t)(kk + r) * N + n0 + c;
#pragma unroll
            for (int j = 0; j < 16; j += 4) {
                const float4 f = *(const float4*)(wp + j);
                const int idx = (r * 64 + c + j) >> 1;
                sW[idx + 0] = f2bf2(f.x, f.y);
                sW[idx + 1] = f2bf2(f.z, f.w);
            }
            if (kk + 32 < K)  // hint next W tile (global_prefetch_b8)
                __builtin_prefetch(W + (size_t)(kk + 32 + r) * N + n0 + c, 0, 1);
        }
        __syncthreads();

        // ---- fragments (u32-packed gathers) ----
        // A (16x32): lane = M (l&15); lanes>=16 take K+8 / K+24 runs
        v8u au;
        {
            const int mA = lane & 15, ks = (lane >> 4) * 8;
            const int b1 = (mA * 32 + ks) >> 1;
            const int b2 = (mA * 32 + 16 + ks) >> 1;
#pragma unroll
            for (int j = 0; j < 4; ++j) { au[j] = sA[b1 + j]; au[4 + j] = sA[b2 + j]; }
        }
        // B (32x16): lane = N (l&15); regs = K runs (K+8 / K+24 for lanes>=16)
        v8u bu;
        {
            const unsigned short* sw16 = (const unsigned short*)sW;
            const int nB = wave * 16 + (lane & 15), ks = (lane >> 4) * 8;
#pragma unroll
            for (int j = 0; j < 8; ++j) {
                const int k0 = ks + 2 * (j & 3) + (j >> 2) * 16;
                bu[j] = (unsigned int)sw16[k0 * 64 + nB] | ((unsigned int)sw16[(k0 + 1) * 64 + nB] << 16);
            }
        }
        acc = __builtin_amdgcn_wmma_f32_16x16x32_bf16(
            false, asbf8(au), false, asbf8(bu), (short)0, acc, false, false);
        __syncthreads();
    }

    // ---- epilogue: lane<16 -> M=r, lane>=16 -> M=8+r; N = lane&15 ----
    const int nl = lane & 15;
    const int n  = n0 + wave * 16 + nl;
#pragma unroll
    for (int r = 0; r < 8; ++r) {
        const int ml = r + ((lane >> 4) * 8);
        const int m  = m0 + ml;
        float v = acc[r] + bias[n];
        if (MODE == 0) {
            ((unsigned short*)out)[(size_t)m * N + n] = f2bf(v);
        } else if (MODE == 1) {
            ((float*)out)[(size_t)m * N + n] = v + res[(size_t)m * N + n];
        } else {
            const int ar = arow(ml);
            ((float*)out)[(size_t)m * N + n] = v + res[(size_t)ar * Dn + (n & (Dn - 1))];
        }
    }
}

// ---------------------------------------------------------------------------
// Banded attention: 1 wave per (b, h, 16-row q tile).
// scores (WMMA) -> mask -> softmax in LDS -> full-row NT writes -> P@V (WMMA)
// ---------------------------------------------------------------------------
__global__ __launch_bounds__(32)
void attn_kernel(const unsigned short* __restrict__ Q,
                 const unsigned short* __restrict__ Kb,
                 const unsigned short* __restrict__ Vb,
                 float* __restrict__ attn_out, float* __restrict__ ctx, int layer)
{
    __shared__ float P[16][292];   // 292 => rows 16B-aligned for b128 LDS reads

    const int lane = threadIdx.x;
    const int q0 = blockIdx.x * 16;
    const int h  = blockIdx.y;
    const int b  = blockIdx.z;
    const float scale = 0.17677669529663687f;   // 1/sqrt(32)

    const size_t qkvBase = ((size_t)b * Ln) * Dn + (size_t)h * DHn;

    // Q fragment (A layout): two 16B runs per lane
    v8u qu;
    {
        const int m = lane & 15, ks = (lane >> 4) * 8;
        const unsigned int* qp =
            (const unsigned int*)(Q + qkvBase + (size_t)(q0 + m) * Dn);
        const uint4 a = *(const uint4*)(qp + (ks >> 1));
        const uint4 c = *(const uint4*)(qp + ((16 + ks) >> 1));
        qu[0] = a.x; qu[1] = a.y; qu[2] = a.z; qu[3] = a.w;
        qu[4] = c.x; qu[5] = c.y; qu[6] = c.z; qu[7] = c.w;
    }
    const v16bf qa = asbf8(qu);

    const int klo      = (q0 >= PASTn) ? (q0 - PASTn) : 0;
    const int ntiles   = (q0 - klo) / 16 + 1;          // 1..17
    const int npad     = (ntiles + 1) & ~1;            // even # of tiles
    const int ncols    = ntiles * 16;
    const int npadcols = npad * 16;

    // ---- scores: S = Q(16x32) @ K^T(32x16) per key tile ----
    for (int t = 0; t < ntiles; ++t) {
        const int kbase = klo + t * 16;
        v16u ku;  // B layout: lane = N (=key), regs = K (=d) runs
#pragma unroll
        for (int j = 0; j < 16; ++j)
            ku[j] = Kb[qkvBase + (size_t)(kbase + (lane & 15)) * Dn + (lane >> 4) * 8 + (j & 7) + (j >> 3) * 16];
        v8f s = {};
        s = __builtin_amdgcn_wmma_f32_16x16x32_bf16(
            false, qa, false, asbf(ku), (short)0, s, false, false);
        const int jcol = kbase + (lane & 15);
#pragma unroll
        for (int r = 0; r < 8; ++r) {
            const int rl = r + ((lane >> 4) * 8);
            const int i  = q0 + rl;
            float v = s[r] * scale;
            if (jcol > i || jcol + PASTn < i) v = -1e9f;
            P[rl][t * 16 + (lane & 15)] = v;
        }
    }
    __syncthreads();

    // ---- softmax over the band (2 lanes per row) ----
    {
        const int row = lane & 15, half = lane >> 4;
        float mx = -3e38f;
        for (int c = half; c < ncols; c += 2) mx = fmaxf(mx, P[row][c]);
        mx = fmaxf(mx, __shfl_xor(mx, 16, 32));
        float sum = 0.f;
        for (int c = half; c < ncols; c += 2) {
            float e = __expf(P[row][c] - mx);
            P[row][c] = e; sum += e;
        }
        sum += __shfl_xor(sum, 16, 32);
        const float rs = 1.0f / sum;
        for (int c = half; c < ncols; c += 2) P[row][c] *= rs;
        for (int c = ncols + half; c < npadcols; c += 2) P[row][c] = 0.f; // pad
    }
    __syncthreads();

    // ---- full-row dense attn output: NT v4f stores (no memset pass) ----
    // Band edges are multiples of 16, c0 steps by 4 => no vector straddle.
    {
        float* base = attn_out + ((size_t)((b * NLn + layer) * Hn + h) * Ln) * Ln;
        for (int r = 0; r < 16; ++r) {
            float* dst = base + (size_t)(q0 + r) * Ln;
#pragma unroll 2
            for (int it = 0; it < Ln / 128; ++it) {      // 8 iters of 32 lanes
                const int c0 = (it * 32 + lane) * 4;
                v4f v = {0.f, 0.f, 0.f, 0.f};
                if (c0 >= klo && c0 < klo + ncols)
                    v = *(const v4f*)&P[r][c0 - klo];
                __builtin_nontemporal_store(v, (v4f*)(dst + c0));
            }
        }
    }

    // ---- ctx = P(16 x band) @ V(band x 32), 32 keys per WMMA pair ----
    v8f acc0 = {}, acc1 = {};
    const int npairs = npad >> 1;
    for (int tp = 0; tp < npairs; ++tp) {
        const int koff = tp * 32;
        v8u pu;  // A layout from P (f32 -> packed bf16)
        {
            const int m = lane & 15, ks = (lane >> 4) * 8;
#pragma unroll
            for (int j = 0; j < 4; ++j) {
                pu[j]     = f2bf2(P[m][koff + ks + 2 * j],      P[m][koff + ks + 2 * j + 1]);
                pu[4 + j] = f2bf2(P[m][koff + 16 + ks + 2 * j], P[m][koff + 16 + ks + 2 * j + 1]);
            }
        }
        const v16bf pa = asbf8(pu);
        const int ksv = (lane >> 4) * 8, dv = lane & 15;   // lane = N (= d)
        v16u vh0, vh1;
#pragma unroll
        for (int j = 0; j < 16; ++j) {
            int key = klo + koff + ksv + (j & 7) + (j >> 3) * 16;
            if (key > q0 + 15) key = q0 + 15;     // clamp pad (P==0 there)
            const unsigned short* vp = Vb + qkvBase + (size_t)key * Dn;
            vh0[j] = vp[dv];
            vh1[j] = vp[16 + dv];
        }
        acc0 = __builtin_amdgcn_wmma_f32_16x16x32_bf16(
            false, pa, false, asbf(vh0), (short)0, acc0, false, false);
        acc1 = __builtin_amdgcn_wmma_f32_16x16x32_bf16(
            false, pa, false, asbf(vh1), (short)0, acc1, false, false);
    }
    {
        const int nl = lane & 15;
#pragma unroll
        for (int r = 0; r < 8; ++r) {
            const int i = q0 + r + ((lane >> 4) * 8);
            float* cp = ctx + ((size_t)b * Ln + i) * Dn + (size_t)h * DHn;
            cp[nl]      = acc0[r];
            cp[16 + nl] = acc1[r];
        }
    }
}

// ---------------------------------------------------------------------------
// Row LayerNorm (D=256): 1 wave per row, 8 rows per 256-thread block
// ---------------------------------------------------------------------------
__global__ __launch_bounds__(256)
void ln_kernel(const float* __restrict__ x, const float* __restrict__ gamma,
               const float* __restrict__ beta, float* __restrict__ y)
{
    const int wave = threadIdx.x >> 5, lane = threadIdx.x & 31;
    const int row  = blockIdx.x * 8 + wave;
    const float* xr = x + (size_t)row * Dn;
    float v[8], s = 0.f;
#pragma unroll
    for (int j = 0; j < 8; ++j) { v[j] = xr[lane + j * 32]; s += v[j]; }
#pragma unroll
    for (int o = 16; o >= 1; o >>= 1) s += __shfl_xor(s, o, 32);
    const float mu = s * (1.0f / Dn);
    float var = 0.f;
#pragma unroll
    for (int j = 0; j < 8; ++j) { const float d = v[j] - mu; var += d * d; }
#pragma unroll
    for (int o = 16; o >= 1; o >>= 1) var += __shfl_xor(var, o, 32);
    const float rstd = __frsqrt_rn(var * (1.0f / Dn) + 1e-5f);
    float* yr = y + (size_t)row * Dn;
#pragma unroll
    for (int j = 0; j < 8; ++j) {
        const int c = lane + j * 32;
        yr[c] = (v[j] - mu) * rstd * gamma[c] + beta[c];
    }
}

// ---------------------------------------------------------------------------
extern "C" void kernel_launch(void* const* d_in, const int* in_sizes, int n_in,
                              void* d_out, int out_size, void* d_ws, size_t ws_size,
                              hipStream_t stream)
{
    (void)in_sizes; (void)n_in; (void)out_size; (void)ws_size;

    const float* inp  = (const float*)d_in[0];
    const float* Wq   = (const float*)d_in[1];
    const float* bq   = (const float*)d_in[2];
    const float* Wk   = (const float*)d_in[3];
    const float* bk   = (const float*)d_in[4];
    const float* Wv   = (const float*)d_in[5];
    const float* bv   = (const float*)d_in[6];
    const float* Wo   = (const float*)d_in[7];
    const float* bo   = (const float*)d_in[8];
    const float* lns  = (const float*)d_in[9];
    const float* lnb  = (const float*)d_in[10];
    const float* Wout = (const float*)d_in[11];
    const float* bout = (const float*)d_in[12];
    // d_in[13] = mask, recomputed analytically in-kernel

    // workspace layout
    char* ws = (char*)d_ws;
    float*          cur = (float*)ws;          ws += (size_t)Mrows * Dn * 4;
    unsigned short* qb  = (unsigned short*)ws; ws += (size_t)Mrows * Dn * 2;
    unsigned short* kb  = (unsigned short*)ws; ws += (size_t)Mrows * Dn * 2;
    unsigned short* vb  = (unsigned short*)ws; ws += (size_t)Mrows * Dn * 2;
    float*          ctx = (float*)ws;          ws += (size_t)Mrows * Dn * 4;
    float*          tmp = (float*)ws;          ws += (size_t)Mrows * Dn * 4;

    float* yout = (float*)d_out;                        // (B, L-PAST, FUT, D)
    float* attn = yout + (size_t)M2rows * NOUT;         // (B, NL*H, L, L)

    (void)hipMemcpyAsync(cur, inp, (size_t)Mrows * Dn * 4,
                         hipMemcpyDeviceToDevice, stream);

    const dim3 gp(Dn / 64, Mrows / 16), bp(128);
    const dim3 ga(Ln / 16, Hn, Bn),     ba(32);
    const dim3 gl(Mrows / 8),           bl(256);

    for (int l = 0; l < NLn; ++l) {
        wmma_gemm<0><<<gp, bp, 0, stream>>>(cur, Wq + (size_t)l * Dn * Dn, bq + l * Dn,
                                            nullptr, qb, Mrows, Dn, Dn);
        wmma_gemm<0><<<gp, bp, 0, stream>>>(cur, Wk + (size_t)l * Dn * Dn, bk + l * Dn,
                                            nullptr, kb, Mrows, Dn, Dn);
        wmma_gemm<0><<<gp, bp, 0, stream>>>(cur, Wv + (size_t)l * Dn * Dn, bv + l * Dn,
                                            nullptr, vb, Mrows, Dn, Dn);
        attn_kernel<<<ga, ba, 0, stream>>>(qb, kb, vb, attn, ctx, l);
        wmma_gemm<1><<<gp, bp, 0, stream>>>(ctx, Wo + (size_t)l * Dn * Dn, bo + l * Dn,
                                            cur, tmp, Mrows, Dn, Dn);
        ln_kernel<<<gl, bl, 0, stream>>>(tmp, lns + l * Dn, lnb + l * Dn, cur);
    }

    wmma_gemm<2><<<dim3(NOUT / 64, M2rows / 16), bp, 0, stream>>>(
        cur, Wout, bout, inp, yout, M2rows, Dn, NOUT);
}
